// EvidentialLoNet_50792283242634
// MI455X (gfx1250) — matrix operand order repaired and checked by reading, not compiled, not run on hardware
//
#include <hip/hip_runtime.h>
#include <hip/hip_bf16.h>
#include <math.h>

typedef _Float16 h16;
typedef __attribute__((ext_vector_type(16))) _Float16 v16h;
typedef __attribute__((ext_vector_type(8)))  _Float16 v8h;
typedef __attribute__((ext_vector_type(8)))  float    v8f;

#define EPS 1e-5f

// ---------------------------------------------------------------------------
// WMMA: D = A(16x32 f16) * B(32x16 f16) + C(16x16 f32), wave32.
// ---------------------------------------------------------------------------
__device__ __forceinline__ v8f wmma16x16x32(v16h a, v16h b, v8f c) {
  return __builtin_amdgcn_wmma_f32_16x16x32_f16(
      false, a, false, b, (short)0, c, false, false);
}

__device__ __forceinline__ v16h cat8(v8h lo, v8h hi) {
  return __builtin_shufflevector(lo, hi, 0, 1, 2, 3, 4, 5, 6, 7,
                                 8, 9, 10, 11, 12, 13, 14, 15);
}

// ---------------------------------------------------------------------------
// y (B,4,45,45) f32 NCHW -> zero-padded f16 NHWC (B,49,49,8):
// halo pad 2, channels 4..7 zero (so conv1 K-octets are full 8-channel runs).
// ---------------------------------------------------------------------------
__global__ void build_ypad(const float* __restrict__ y, h16* __restrict__ d, int Batch) {
  const size_t i = (size_t)blockIdx.x * blockDim.x + threadIdx.x;
  const size_t total = (size_t)Batch * 49 * 49 * 8;
  if (i >= total) return;
  const int c  = (int)(i % 8);
  size_t t = i / 8;
  const int px = (int)(t % 49); t /= 49;
  const int py = (int)(t % 49);
  const int b  = (int)(t / 49);
  h16 v = (h16)0;
  if (c < 4 && py >= 2 && py < 47 && px >= 2 && px < 47)
    v = (h16)y[(((size_t)b * 4 + c) * 45 + (py - 2)) * 45 + (px - 2)];
  d[i] = v;
}

__global__ void fill_h16(h16* __restrict__ d, size_t n) {
  const size_t i = (size_t)blockIdx.x * blockDim.x + threadIdx.x;
  if (i < n) d[i] = (h16)0;
}

// Conv weights (Cout,Cin,KS,KS) f32 -> f16 (Cout,Kp) with K order (ky,kx,c),
// c padded to Cp, K padded to Kp (zeros).
__global__ void cvt_w_conv(const float* __restrict__ s, h16* __restrict__ d,
                           int N, int KS, int Cin, int Cp, int Kp) {
  const size_t i = (size_t)blockIdx.x * blockDim.x + threadIdx.x;
  if (i >= (size_t)N * Kp) return;
  const int k = (int)(i % Kp);
  const int n = (int)(i / Kp);
  const int tapi = k / Cp, c = k - (k / Cp) * Cp;
  h16 v = (h16)0;
  if (tapi < KS * KS && c < Cin) {
    const int ky = tapi / KS, kx = tapi - (tapi / KS) * KS;
    v = (h16)s[(((size_t)n * Cin + c) * KS + ky) * KS + kx];
  }
  d[i] = v;
}

// fc1 weights (512,2304) with K permuted to match conv3's NHWC output:
// k_dst = pos*256 + c  <->  k_src = c*9 + pos.
__global__ void cvt_w_fc1(const float* __restrict__ s, h16* __restrict__ d) {
  const size_t i = (size_t)blockIdx.x * blockDim.x + threadIdx.x;
  if (i >= (size_t)512 * 2304) return;
  const int k = (int)(i % 2304);
  const int n = (int)(i / 2304);
  const int pos = k / 256, c = k - (k / 256) * 256;
  d[i] = (h16)s[(size_t)n * 2304 + c * 9 + pos];
}

// Plain (N,K) f32 -> f16 (K multiple of 32 already).
__global__ void cvt_w(const float* __restrict__ s, h16* __restrict__ d, size_t n) {
  const size_t i = (size_t)blockIdx.x * blockDim.x + threadIdx.x;
  if (i < n) d[i] = (h16)s[i];
}

// ---------------------------------------------------------------------------
// Implicit-im2col conv + bias + BN + ReLU on the WMMA pipe, NHWC activations.
//   GEMM: M = B*Hout*Hout, N = Cout, K = KS*KS*Cp (padded to Kp).
//   K order (ky,kx,c): every 8-slot A octet = 8 consecutive channels of one
//   tap -> one aligned v8h load. Per-octet offsets live in a tiny LDS table.
//   Unsigned 32-bit lane offsets (SADDR-form addressing) + manual double
//   buffering so step k+1's loads are in flight across step k's 4 WMMAs.
// ---------------------------------------------------------------------------
template<int Cp, int KS, int Str, int Hp, int Hout, int Cout, int Kp,
         int OutHp, int OutPad, int OutC>
__global__ __launch_bounds__(128)
void conv_bn_relu(const h16* __restrict__ in, const h16* __restrict__ w,
                  const float* __restrict__ cbias,
                  const float* __restrict__ bg, const float* __restrict__ bb,
                  const float* __restrict__ bm, const float* __restrict__ bv,
                  h16* __restrict__ out, int Batch) {
  constexpr int MT   = 4;
  constexpr int NOCT = Kp / 8;
  __shared__ unsigned tapOct[NOCT];
  for (int o = threadIdx.x; o < NOCT; o += 128) {
    const int k    = o * 8;
    const int tapi = k / Cp;
    const int c0   = k - tapi * Cp;
    const int ky   = tapi / KS, kx = tapi - (tapi / KS) * KS;
    tapOct[o] = (unsigned)((ky * Hp + kx) * Cp + c0);  // octet element offset
  }
  __syncthreads();

  const int wave = threadIdx.x >> 5;
  const int lane = threadIdx.x & 31;
  const int HW   = Hout * Hout;
  const int Msup = (Batch * HW) >> 6;
  const int tile = blockIdx.x * 4 + wave;      // wave-uniform
  if (tile >= Msup * (Cout >> 4)) return;
  const int mts = tile % Msup, nt = tile / Msup;
  const int rowBase = mts << 6;
  const int n0      = nt << 4;

  // Unsigned per-lane input offsets (NHWC padded coords) -> SADDR loads.
  unsigned bOff[MT];
#pragma unroll
  for (int s = 0; s < MT; ++s) {
    const int m  = rowBase + (s << 4) + (lane & 15);
    const int b_ = m / HW;
    const int rm = m - b_ * HW;
    const int oy = rm / Hout, ox = rm - (rm / Hout) * Hout;
    bOff[s] = (unsigned)(((b_ * Hp + oy * Str) * Hp + ox * Str) * Cp);
  }
  const int      aob  = (lane & 16) ? 1 : 0;   // octet shift
  const unsigned wOff = (unsigned)((n0 + (lane & 15)) * Kp + ((lane & 16) ? 8 : 0));

  v8f acc[MT] = {};
  v8h alo[MT], ahi[MT];
  v16h bcur;
  {
    const unsigned o0 = tapOct[aob], o1 = tapOct[aob + 2];
    bcur = cat8(*(const v8h*)(w + wOff), *(const v8h*)(w + (wOff + 16u)));
#pragma unroll
    for (int s = 0; s < MT; ++s) {
      alo[s] = *(const v8h*)(in + (bOff[s] + o0));
      ahi[s] = *(const v8h*)(in + (bOff[s] + o1));
    }
  }
  for (int k0 = 0; k0 < Kp - 32; k0 += 32) {
    const int kn = k0 + 32;
    // Prefetch step k+1 before consuming step k.
    const v16h bnext = cat8(*(const v8h*)(w + (wOff + (unsigned)kn)), *(const v8h*)(w + (wOff + (unsigned)kn + 16u)));
    const unsigned no0 = tapOct[(kn >> 3) + aob];
    const unsigned no1 = tapOct[(kn >> 3) + aob + 2];
    v8h nlo[MT], nhi[MT];
#pragma unroll
    for (int s = 0; s < MT; ++s) {
      nlo[s] = *(const v8h*)(in + (bOff[s] + no0));
      nhi[s] = *(const v8h*)(in + (bOff[s] + no1));
    }
#pragma unroll
    for (int s = 0; s < MT; ++s)
      acc[s] = wmma16x16x32(cat8(alo[s], ahi[s]), bcur, acc[s]);
    bcur = bnext;
#pragma unroll
    for (int s = 0; s < MT; ++s) { alo[s] = nlo[s]; ahi[s] = nhi[s]; }
  }
#pragma unroll
  for (int s = 0; s < MT; ++s)
    acc[s] = wmma16x16x32(cat8(alo[s], ahi[s]), bcur, acc[s]);

  // Epilog: (conv + bias - mean)*g/sqrt(v+eps) + beta, ReLU, NHWC interior.
  const int   ch = n0 + (lane & 15);
  const float sc = bg[ch] * rsqrtf(bv[ch] + EPS);
  const float sh = (cbias[ch] - bm[ch]) * sc + bb[ch];
  const int rb = (lane & 16) ? 8 : 0;
#pragma unroll
  for (int s = 0; s < MT; ++s) {
#pragma unroll
    for (int r = 0; r < 8; ++r) {
      const int m  = rowBase + (s << 4) + rb + r;
      float val = acc[s][r] * sc + sh;
      val = val > 0.f ? val : 0.f;
      const int b_ = m / HW;
      const int rm = m - b_ * HW;
      const int oy = rm / Hout, ox = rm - (rm / Hout) * Hout;
      out[(((size_t)b_ * OutHp + oy + OutPad) * OutHp + (ox + OutPad)) * OutC + ch]
          = (h16)val;   // channel-contiguous across lanes (coalesced)
    }
  }
}

// ---------------------------------------------------------------------------
// FC GEMM: out = A(MxK f16) * W(N,K f16)^T + bias. 64x16 per wave; K is
// compile-time; unsigned offsets + double buffering as in the conv kernel.
// ---------------------------------------------------------------------------
template<bool RELU, bool OUTF32, int K>
__global__ __launch_bounds__(128)
void fc_wmma(const h16* __restrict__ A, const h16* __restrict__ W,
             const float* __restrict__ bias,
             h16* __restrict__ outH, float* __restrict__ outF,
             int M, int N) {
  constexpr int MT = 4;
  const int wave = threadIdx.x >> 5, lane = threadIdx.x & 31;
  const int Msup = M >> 6;
  const int tile = blockIdx.x * 4 + wave;
  if (tile >= Msup * (N >> 4)) return;
  const int mts = tile % Msup, nt = tile / Msup;
  const int rowBase = mts << 6, n0 = nt << 4;
  const unsigned akb = (lane & 16) ? 8u : 0u;

  unsigned aOff[MT];
#pragma unroll
  for (int s = 0; s < MT; ++s)
    aOff[s] = (unsigned)((rowBase + (s << 4) + (lane & 15)) * K) + akb;
  const unsigned wOff = (unsigned)((n0 + (lane & 15)) * K + ((lane & 16) ? 8 : 0));

  v8f acc[MT] = {};
  v8h alo[MT], ahi[MT];
  v16h bcur;
  {
    bcur = cat8(*(const v8h*)(W + wOff), *(const v8h*)(W + (wOff + 16u)));
#pragma unroll
    for (int s = 0; s < MT; ++s) {
      alo[s] = *(const v8h*)(A + aOff[s]);
      ahi[s] = *(const v8h*)(A + (aOff[s] + 16u));
    }
  }
  for (int k0 = 0; k0 < K - 32; k0 += 32) {
    const unsigned kn = (unsigned)(k0 + 32);
    const v16h bnext = cat8(*(const v8h*)(W + (wOff + kn)), *(const v8h*)(W + (wOff + kn + 16u)));
    v8h nlo[MT], nhi[MT];
#pragma unroll
    for (int s = 0; s < MT; ++s) {
      nlo[s] = *(const v8h*)(A + (aOff[s] + kn));
      nhi[s] = *(const v8h*)(A + (aOff[s] + kn + 16u));
    }
#pragma unroll
    for (int s = 0; s < MT; ++s)
      acc[s] = wmma16x16x32(cat8(alo[s], ahi[s]), bcur, acc[s]);
    bcur = bnext;
#pragma unroll
    for (int s = 0; s < MT; ++s) { alo[s] = nlo[s]; ahi[s] = nhi[s]; }
  }
#pragma unroll
  for (int s = 0; s < MT; ++s)
    acc[s] = wmma16x16x32(cat8(alo[s], ahi[s]), bcur, acc[s]);

  const int   n   = n0 + (lane & 15);
  const float bsv = bias[n];
  const int rb = (lane & 16) ? 8 : 0;
#pragma unroll
  for (int s = 0; s < MT; ++s) {
#pragma unroll
    for (int r = 0; r < 8; ++r) {
      const int m = rowBase + (s << 4) + rb + r;
      float v = acc[s][r] + bsv;
      if (RELU) v = v > 0.f ? v : 0.f;
      if (OUTF32) outF[(size_t)m * N + n] = v;
      else        outH[(size_t)m * N + n] = (h16)v;
    }
  }
}

// ---------------------------------------------------------------------------
// GRU branch (2 layers, H=16, T=14) + bn + classifier, 1 thread / sample,
// weights in LDS. Sequential & tiny -> VALU by design, not WMMA-shaped.
// ---------------------------------------------------------------------------
__device__ __forceinline__ float sigm(float x) { return 1.f / (1.f + expf(-x)); }

__global__ __launch_bounds__(256)
void gru_head(const float* __restrict__ x,
              const float* w0i, const float* w0h, const float* b0i, const float* b0h,
              const float* w1i, const float* w1h, const float* b1i, const float* b1h,
              const float* bng, const float* bnb, const float* bnm, const float* bnv,
              const float* clsw, const float* clsb,
              float* __restrict__ gf, int Batch) {
  __shared__ float s0i[48 * 4], s0h[48 * 16], s1i[48 * 16], s1h[48 * 16];
  __shared__ float sb0i[48], sb0h[48], sb1i[48], sb1h[48];
  __shared__ float sclw[64 * 16], sclb[64];
  for (int i = threadIdx.x; i < 48 * 4; i += 256) s0i[i] = w0i[i];
  for (int i = threadIdx.x; i < 48 * 16; i += 256) {
    s0h[i] = w0h[i]; s1i[i] = w1i[i]; s1h[i] = w1h[i];
  }
  for (int i = threadIdx.x; i < 48; i += 256) {
    sb0i[i] = b0i[i]; sb0h[i] = b0h[i]; sb1i[i] = b1i[i]; sb1h[i] = b1h[i];
  }
  for (int i = threadIdx.x; i < 64 * 16; i += 256) sclw[i] = clsw[i];
  for (int i = threadIdx.x; i < 64; i += 256) sclb[i] = clsb[i];
  __syncthreads();

  const int b = blockIdx.x * 256 + threadIdx.x;
  if (b >= Batch) return;

  float h0[16], h1[16], hn[16];
#pragma unroll
  for (int i = 0; i < 16; ++i) { h0[i] = 0.f; h1[i] = 0.f; }
  const float* xb = x + (size_t)b * 14 * 4;

  for (int t = 0; t < 14; ++t) {
    float xt[4];
#pragma unroll
    for (int i = 0; i < 4; ++i) xt[i] = xb[t * 4 + i];
#pragma unroll
    for (int k = 0; k < 16; ++k) {        // layer 0 (input dim 4)
      float xr = sb0i[k], xz = sb0i[16 + k], xn = sb0i[32 + k];
      float hr = sb0h[k], hz = sb0h[16 + k], hgn = sb0h[32 + k];
#pragma unroll
      for (int i = 0; i < 4; ++i) {
        xr += s0i[k * 4 + i] * xt[i];
        xz += s0i[(16 + k) * 4 + i] * xt[i];
        xn += s0i[(32 + k) * 4 + i] * xt[i];
      }
#pragma unroll
      for (int i = 0; i < 16; ++i) {
        hr  += s0h[k * 16 + i] * h0[i];
        hz  += s0h[(16 + k) * 16 + i] * h0[i];
        hgn += s0h[(32 + k) * 16 + i] * h0[i];
      }
      const float r = sigm(xr + hr), z = sigm(xz + hz);
      const float n = tanhf(xn + r * hgn);
      hn[k] = (1.f - z) * n + z * h0[k];
    }
#pragma unroll
    for (int k = 0; k < 16; ++k) h0[k] = hn[k];
#pragma unroll
    for (int k = 0; k < 16; ++k) {        // layer 1 (input = h0)
      float xr = sb1i[k], xz = sb1i[16 + k], xn = sb1i[32 + k];
      float hr = sb1h[k], hz = sb1h[16 + k], hgn = sb1h[32 + k];
#pragma unroll
      for (int i = 0; i < 16; ++i) {
        xr += s1i[k * 16 + i] * h0[i];
        xz += s1i[(16 + k) * 16 + i] * h0[i];
        xn += s1i[(32 + k) * 16 + i] * h0[i];
        hr  += s1h[k * 16 + i] * h1[i];
        hz  += s1h[(16 + k) * 16 + i] * h1[i];
        hgn += s1h[(32 + k) * 16 + i] * h1[i];
      }
      const float r = sigm(xr + hr), z = sigm(xz + hz);
      const float n = tanhf(xn + r * hgn);
      hn[k] = (1.f - z) * n + z * h1[k];
    }
#pragma unroll
    for (int k = 0; k < 16; ++k) h1[k] = hn[k];
  }

  float hb[16];
#pragma unroll
  for (int k = 0; k < 16; ++k)
    hb[k] = (h1[k] - bnm[k]) * (bng[k] * rsqrtf(bnv[k] + EPS)) + bnb[k];
#pragma unroll 4
  for (int o = 0; o < 64; ++o) {
    float s = sclb[o];
#pragma unroll
    for (int k = 0; k < 16; ++k) s += sclw[o * 16 + k] * hb[k];
    gf[(size_t)b * 64 + o] = s;
  }
}

// ---------------------------------------------------------------------------
// Evidential heads + Dempster fusion, 1 thread / sample.
// ---------------------------------------------------------------------------
__device__ __forceinline__ float softplusf(float x) {
  return x > 0.f ? x + log1pf(expf(-x)) : log1pf(expf(x));
}

__device__ void head_eval(const float* __restrict__ f,
                          const float* w1, const float* b1,
                          const float* bg, const float* bb,
                          const float* bm, const float* bv,
                          const float* w2, const float* b2, float* a) {
  float h[32];
#pragma unroll 4
  for (int j = 0; j < 32; ++j) {
    float s = b1[j];
#pragma unroll 8
    for (int i = 0; i < 64; ++i) s += w1[j * 64 + i] * f[i];
    s = s > 0.f ? s : 0.f;
    h[j] = (s - bm[j]) * (bg[j] * rsqrtf(bv[j] + EPS)) + bb[j];
  }
#pragma unroll
  for (int c = 0; c < 4; ++c) {
    float s = b2[c];
#pragma unroll
    for (int j = 0; j < 32; ++j) s += w2[c * 32 + j] * h[j];
    a[c] = softplusf(s) + 1.f;
  }
}

__global__ __launch_bounds__(256)
void head_fusion(const float* __restrict__ gf, const float* __restrict__ cf,
                 const float* gw1, const float* gb1,
                 const float* gbg, const float* gbb, const float* gbm, const float* gbv,
                 const float* gw2, const float* gb2,
                 const float* cw1, const float* cb1,
                 const float* cbg, const float* cbb, const float* cbm, const float* cbv,
                 const float* cw2, const float* cb2,
                 float* __restrict__ out, int Batch) {
  const int b = blockIdx.x * 256 + threadIdx.x;
  if (b >= Batch) return;
  float a0[4], a1[4];
  head_eval(gf + (size_t)b * 64, gw1, gb1, gbg, gbb, gbm, gbv, gw2, gb2, a0);
  head_eval(cf + (size_t)b * 64, cw1, cb1, cbg, cbb, cbm, cbv, cw2, cb2, a1);

  float S0 = 0.f, S1 = 0.f;
#pragma unroll
  for (int c = 0; c < 4; ++c) { S0 += a0[c]; S1 += a1[c]; }
  const float u0 = 4.f / S0, u1 = 4.f / S1;
  const float q0 = 1.f - u0, q1 = 1.f - u1;
  float c0[4], c1[4], cc = 0.f;
#pragma unroll
  for (int c = 0; c < 4; ++c) {
    c0[c] = (a0[c] - 1.f) / S0;
    c1[c] = (a1[c] - 1.f) / S1;
    cc += c0[c] * c1[c];
  }
  const float denom = cc + q0 * q0 + q1 * q1 + q0 * q1;
  const float u_tot = q0 * q1 / denom;
  const float S_tot = 4.f / u_tot;
#pragma unroll
  for (int c = 0; c < 4; ++c) {
    const float ct = (c0[c] * c1[c] + c0[c] * q0 + c1[c] * q1) / denom;
    out[(size_t)b * 4 + c] = ct * S_tot;                    // evidence_after
    out[(size_t)Batch * 4 + (size_t)b * 4 + c] = a0[c];     // a_gru
    out[(size_t)Batch * 8 + (size_t)b * 4 + c] = a1[c];     // a_cnn
  }
}

// ---------------------------------------------------------------------------
// Launch. Inputs in recursive insertion order of setup_inputs():
//   0:x 1:y 2:conv1_w 3:conv1_b 4..7:bn1(g,b,m,v) 8:conv2_w 9:conv2_b
//   10..13:bn2 14:conv3_w 15:conv3_b 16..19:bn3 20:fc1_w 21:fc1_b 22:fc2_w
//   23:fc2_b 24:fc3_w 25:fc3_b 26:gru0_wih 27:gru0_whh 28:gru0_bih 29:gru0_bhh
//   30:gru1_wih 31:gru1_whh 32:gru1_bih 33:gru1_bhh 34..37:bn_gru 38:cls_w
//   39:cls_b 40:hg_w1 41:hg_b1 42..45:hg_bn 46:hg_w2 47:hg_b2 48:hc_w1
//   49:hc_b1 50..53:hc_bn 54:hc_w2 55:hc_b2
// ---------------------------------------------------------------------------
extern "C" void kernel_launch(void* const* d_in, const int* in_sizes, int n_in,
                              void* d_out, int out_size, void* d_ws, size_t ws_size,
                              hipStream_t stream) {
  auto F = [&](int i) { return (const float*)d_in[i]; };
  const float* x = F(0);
  const float* y = F(1);
  const int B = in_sizes[0] / (14 * 4);   // 4096

  size_t off = 0;
  auto alloc = [&](size_t bytes) {
    void* p = (char*)d_ws + off;
    off += (bytes + 255) & ~(size_t)255;
    return p;
  };
  h16* ypad = (h16*)alloc((size_t)B * 49 * 49 * 8 * 2);   // NHWC, Cp=8
  h16* w1h  = (h16*)alloc((size_t)64 * 1824 * 2);
  h16* w2h  = (h16*)alloc((size_t)128 * 1600 * 2);
  h16* w3h  = (h16*)alloc((size_t)256 * 3200 * 2);
  h16* f1h  = (h16*)alloc((size_t)512 * 2304 * 2);
  h16* f2h  = (h16*)alloc((size_t)128 * 512 * 2);
  h16* f3h  = (h16*)alloc((size_t)64 * 128 * 2);
  h16* h1p  = (h16*)alloc((size_t)B * 16 * 16 * 64 * 2);  // NHWC padded
  h16* h2p  = (h16*)alloc((size_t)B * 10 * 10 * 128 * 2); // NHWC padded
  h16* h3   = (h16*)alloc((size_t)B * 2304 * 2);          // (B, pos*256+c)
  h16* o1   = (h16*)alloc((size_t)B * 512 * 2);
  h16* o2   = (h16*)alloc((size_t)B * 128 * 2);
  float* cf = (float*)alloc((size_t)B * 64 * 4);
  float* gf = (float*)alloc((size_t)B * 64 * 4);
  (void)ws_size; (void)n_in;

  auto gb = [](size_t n) { return (unsigned)((n + 255) / 256); };

  // Stage inputs / weights: NHWC zero-padding + (ky,kx,c) K-permute + f16.
  {
    const size_t nyp = (size_t)B * 49 * 49 * 8;
    build_ypad<<<gb(nyp), 256, 0, stream>>>(y, ypad, B);
    const size_t n1 = (size_t)B * 16 * 16 * 64, n2 = (size_t)B * 10 * 10 * 128;
    fill_h16<<<gb(n1), 256, 0, stream>>>(h1p, n1);   // zero halos
    fill_h16<<<gb(n2), 256, 0, stream>>>(h2p, n2);
    cvt_w_conv<<<gb((size_t)64 * 1824), 256, 0, stream>>>(F(2), w1h, 64, 15, 4, 8, 1824);
    cvt_w_conv<<<gb((size_t)128 * 1600), 256, 0, stream>>>(F(8), w2h, 128, 5, 64, 64, 1600);
    cvt_w_conv<<<gb((size_t)256 * 3200), 256, 0, stream>>>(F(14), w3h, 256, 5, 128, 128, 3200);
    cvt_w_fc1<<<gb((size_t)512 * 2304), 256, 0, stream>>>(F(20), f1h);
    cvt_w<<<gb((size_t)128 * 512), 256, 0, stream>>>(F(22), f2h, (size_t)128 * 512);
    cvt_w<<<gb((size_t)64 * 128), 256, 0, stream>>>(F(24), f3h, (size_t)64 * 128);
  }

  auto gemm_blocks = [](int M, int N) { return (unsigned)(((M >> 6) * (N >> 4) + 3) / 4); };

  // CNN branch (WMMA pipe)
  conv_bn_relu<8, 15, 3, 49, 12, 64, 1824, 16, 2, 64>
      <<<gemm_blocks(B * 144, 64), 128, 0, stream>>>(
          ypad, w1h, F(3), F(4), F(5), F(6), F(7), h1p, B);
  conv_bn_relu<64, 5, 2, 16, 6, 128, 1600, 10, 2, 128>
      <<<gemm_blocks(B * 36, 128), 128, 0, stream>>>(
          h1p, w2h, F(9), F(10), F(11), F(12), F(13), h2p, B);
  conv_bn_relu<128, 5, 2, 10, 3, 256, 3200, 3, 0, 256>
      <<<gemm_blocks(B * 9, 256), 128, 0, stream>>>(
          h2p, w3h, F(15), F(16), F(17), F(18), F(19), h3, B);
  fc_wmma<true,  false, 2304><<<gemm_blocks(B, 512), 128, 0, stream>>>(h3, f1h, F(21), o1, nullptr, B, 512);
  fc_wmma<false, false, 512 ><<<gemm_blocks(B, 128), 128, 0, stream>>>(o1, f2h, F(23), o2, nullptr, B, 128);
  fc_wmma<false, true,  128 ><<<gemm_blocks(B, 64),  128, 0, stream>>>(o2, f3h, F(25), nullptr, cf, B, 64);

  // GRU branch + classifier
  gru_head<<<(B + 255) / 256, 256, 0, stream>>>(
      x, F(26), F(27), F(28), F(29), F(30), F(31), F(32), F(33),
      F(34), F(35), F(36), F(37), F(38), F(39), gf, B);

  // Heads + evidential fusion -> d_out (evidence | a_gru | a_cnn)
  head_fusion<<<(B + 255) / 256, 256, 0, stream>>>(
      gf, cf,
      F(40), F(41), F(42), F(43), F(44), F(45), F(46), F(47),
      F(48), F(49), F(50), F(51), F(52), F(53), F(54), F(55),
      (float*)d_out, B);
  (void)out_size;
}
